// Attention_75093208203309
// MI455X (gfx1250) — matrix operand order repaired and checked by reading, not compiled, not run on hardware
//
#include <hip/hip_runtime.h>
#include <hip/hip_bf16.h>

#define NTOK 4096
#define DDIM 1024

typedef __bf16 bf16_t;
typedef __attribute__((ext_vector_type(16))) __bf16 bf16x16;
typedef __attribute__((ext_vector_type(8)))  float  f32x8;
typedef uint32_t u32x4 __attribute__((ext_vector_type(4)));
typedef uint32_t u32x8 __attribute__((ext_vector_type(8)));

union ABFrag { bf16x16 v; uint4 q[2]; };

static __device__ __forceinline__ f32x8 wmma_bf16(bf16x16 a, bf16x16 b, f32x8 c) {
  return __builtin_amdgcn_wmma_f32_16x16x32_bf16(false, a, false, b, (short)0, c,
                                                 false, false);
}

// ---------------------------------------------------------------------------
// Tensor Data Mover: DMA a 16-row x 2048-byte tile (row stride 2048B) from
// global memory into LDS. D# per CDNA5 ISA:
//   group0: count=1 | lds_addr | global_addr[56:0] | type=2
//   group1: data_size=3 (8B units), tensor_dim0=256, tensor_dim1=16,
//           tile_dim0=256, tile_dim1=16, tensor_dim0_stride=256
// 2-D tensor -> descriptor groups 2/3 omitted (VADDR2/3 = NULL).
// Issue from ONE wave only; completion tracked by TENSORcnt.
// ---------------------------------------------------------------------------
static __device__ __forceinline__ void tdm_load_tile_16x2048(const void* gsrc,
                                                             void* ldst) {
  const uint64_t ga = (uint64_t)(uintptr_t)gsrc;
  const uint32_t la = (uint32_t)(uintptr_t)ldst;  // LDS aperture: addr[31:0]
  u32x4 g0;
  g0[0] = 1u;                                               // count = 1
  g0[1] = la;                                               // lds_addr
  g0[2] = (uint32_t)ga;                                     // global_addr[31:0]
  g0[3] = (uint32_t)((ga >> 32) & 0x01FFFFFFu) | (2u << 30);// addr[56:32]|type=2
  u32x8 g1;
  g1[0] = 3u << 16;      // workgroup_mask=0, data_size=3 (8 bytes)
  g1[1] = 256u << 16;    // tensor_dim0[15:0] = 256 (bits 79:48 -> low half)
  g1[2] = 16u << 16;     // tensor_dim0[31:16]=0, tensor_dim1 = 16
  g1[3] = 256u << 16;    // tensor_dim1[31:16]=0, tile_dim0 = 256
  g1[4] = 16u;           // tile_dim1 = 16, tile_dim2 = 0
  g1[5] = 256u;          // tensor_dim0_stride[31:0] = 256 (8B units = 2048B)
  g1[6] = 0u;            // dim0_stride[47:32]=0, dim1_stride[15:0]=0
  g1[7] = 0u;            // dim1_stride[47:16]=0
  asm volatile("tensor_load_to_lds %0, %1" :: "s"(g0), "s"(g1) : "memory");
}

// ---------------------------------------------------------------------------
// fp32 -> bf16 cast (vectorized x4)
// ---------------------------------------------------------------------------
__global__ void cast_f32_bf16_v4(const float* __restrict__ in,
                                 bf16_t* __restrict__ out, int n4) {
  int i = blockIdx.x * blockDim.x + threadIdx.x;
  if (i < n4) {
    float4 f = ((const float4*)in)[i];
    union { uint2 u; bf16_t h[4]; } pk;
    pk.h[0] = (bf16_t)f.x; pk.h[1] = (bf16_t)f.y;
    pk.h[2] = (bf16_t)f.z; pk.h[3] = (bf16_t)f.w;
    ((uint2*)out)[i] = pk.u;
  }
}

// fp32 [rows x cols] -> bf16 transposed [cols x rows]
__global__ void cast_transpose_bf16(const float* __restrict__ in,
                                    bf16_t* __restrict__ out, int rows, int cols) {
  int idx = blockIdx.x * blockDim.x + threadIdx.x;
  if (idx < rows * cols) {
    int r = idx / cols, c = idx % cols;
    out[(size_t)c * rows + r] = (bf16_t)in[idx];
  }
}

// ---------------------------------------------------------------------------
// GEMM: out[i,n] = sum_k A[i,k] * B[n,k] (+ bias[n])
// A: [NTOK x DDIM] bf16 row-major. B: [DDIM x DDIM] bf16, indexed B[n*DDIM+k]
// transposedOut==0: out bf16 row-major; ==1: out bf16 [DDIM x NTOK] (V^T).
// Block = 256 threads = 8 waves; wave w computes one 16x16 tile; 32 WMMAs.
// ---------------------------------------------------------------------------
__global__ __launch_bounds__(256)
void gemm16_bf16(const bf16_t* __restrict__ A, const bf16_t* __restrict__ B,
                 const float* __restrict__ bias, bf16_t* __restrict__ out,
                 int transposedOut) {
  __shared__ __align__(16) bf16_t sA[16 * DDIM];
  const int tid  = threadIdx.x;
  const int wave = tid >> 5, lane = tid & 31;
  const int half = lane >> 4, lcol = lane & 15;
  const int i0 = blockIdx.x * 16;
  const int n0 = blockIdx.y * 128 + wave * 16;

  // TDM: stage the 16 A-rows (16 x 2048B) for this block into LDS
  if (wave == 0) {
    tdm_load_tile_16x2048(A + (size_t)i0 * DDIM, sA);
    __builtin_amdgcn_s_wait_tensorcnt(0);
  }
  __syncthreads();

  float bv = (bias != nullptr) ? bias[n0 + lcol] : 0.0f;
  f32x8 acc = {bv, bv, bv, bv, bv, bv, bv, bv};

  for (int kk = 0; kk < DDIM / 32; ++kk) {
    const int k = kk * 32;
    ABFrag a, b;
    // A fragment: lane<16 -> row lcol, K = k+[0..7] & k+[16..23]; lane>=16 -> +8
    const bf16_t* ap = &sA[lcol * DDIM + k + half * 8];
    a.q[0] = *(const uint4*)(ap);
    a.q[1] = *(const uint4*)(ap + 16);
    // B fragment: lane<16 -> col n0+lcol, K = k+[0..7] & k+[16..23]; lane>=16 -> +8
    const bf16_t* bp = B + (size_t)(n0 + lcol) * DDIM + k + half * 8;
    b.q[0] = *(const uint4*)(bp);
    b.q[1] = *(const uint4*)(bp + 16);
    acc = wmma_bf16(a.v, b.v, acc);
  }

  if (!transposedOut) {
#pragma unroll
    for (int r = 0; r < 8; ++r) {
      const int M = r + half * 8;
      out[(size_t)(i0 + M) * DDIM + n0 + lcol] = (bf16_t)acc[r];
    }
  } else {
    union { uint4 u; bf16_t h[8]; } pk;
#pragma unroll
    for (int r = 0; r < 8; ++r) pk.h[r] = (bf16_t)acc[r];
    *(uint4*)(out + (size_t)(n0 + lcol) * NTOK + i0 + half * 8) = pk.u;
  }
}

// ---------------------------------------------------------------------------
// Flash attention with straight-through connection mask (see round-1 notes).
// ---------------------------------------------------------------------------
__global__ __launch_bounds__(256)
void flash_attn_st(const bf16_t* __restrict__ Qb, const bf16_t* __restrict__ Kb,
                   const bf16_t* __restrict__ Vt, const bf16_t* __restrict__ Cb,
                   const float* __restrict__ am, const float* __restrict__ lmsk,
                   const float* __restrict__ bmask, const float* __restrict__ x,
                   const float* __restrict__ biasp, float* __restrict__ out) {
  __shared__ __align__(16) bf16_t sQ[16 * DDIM];   // 32 KB
  __shared__ __align__(16) bf16_t sP[16 * 128];    // 4 KB
  __shared__ float wred[8][16];
  __shared__ float rowmax[16], rowsum[16], rowscale[16];

  const int tid  = threadIdx.x;
  const int wave = tid >> 5, lane = tid & 31;
  const int half = lane >> 4, lcol = lane & 15;
  const int i0 = blockIdx.x * 16;
  const float cbias = biasp[0];
  const float invs  = 0.03125f;  // 1/sqrt(1024)

  // TDM: stage this block's 16 Q-rows into LDS
  if (wave == 0) {
    tdm_load_tile_16x2048(Qb + (size_t)i0 * DDIM, sQ);
    __builtin_amdgcn_s_wait_tensorcnt(0);
  }
  if (tid < 16) { rowmax[tid] = -3.0e38f; rowsum[tid] = 0.0f; }
  __syncthreads();

  const f32x8 zero = {0, 0, 0, 0, 0, 0, 0, 0};
  f32x8 acc[8];
#pragma unroll
  for (int dt = 0; dt < 8; ++dt) acc[dt] = zero;

  for (int jb = 0; jb < NTOK / 128; ++jb) {
    const int j0 = jb * 128 + wave * 16;

    // prefetch next j-block's mask elements into L2 (global_prefetch_b8)
    if (jb + 1 < NTOK / 128) {
      size_t poff = (size_t)(i0 + half * 8) * NTOK + (jb + 1) * 128 + wave * 16 + lcol;
      __builtin_prefetch(am + poff, 0, 1);
      __builtin_prefetch(lmsk + poff, 0, 1);
    }

    // ---- phase 1: S = Q K^T tile, CL = (Q conn) K^T tile -------------------
    f32x8 S = zero, CL = zero;
    for (int kk = 0; kk < DDIM / 32; ++kk) {
      const int k = kk * 32;
      ABFrag aq, ac, bk;
      const bf16_t* qp = &sQ[lcol * DDIM + k + half * 8];
      aq.q[0] = *(const uint4*)(qp);
      aq.q[1] = *(const uint4*)(qp + 16);
      const bf16_t* cp = Cb + (size_t)(i0 + lcol) * DDIM + k + half * 8;
      ac.q[0] = *(const uint4*)(cp);
      ac.q[1] = *(const uint4*)(cp + 16);
      const bf16_t* kp = Kb + (size_t)(j0 + lcol) * DDIM + k + half * 8;
      bk.q[0] = *(const uint4*)(kp);
      bk.q[1] = *(const uint4*)(kp + 16);
      S  = wmma_bf16(aq.v, bk.v, S);
      CL = wmma_bf16(ac.v, bk.v, CL);
    }

    // masked logits
    float l[8];
#pragma unroll
    for (int r = 0; r < 8; ++r) {
      const int M = r + half * 8;
      const size_t off = (size_t)(i0 + M) * NTOK + jb * 128 + wave * 16 + lcol;
      const float amv = am[off];
      const float lmv = lmsk[off];
      const float st  = (CL[r] + cbias) > 0.0f ? 1.0f : 0.0f;  // sigmoid>0.5
      const float msk = amv + st * lmv;
      l[r] = S[r] * invs - 10000.0f * (1.0f - msk);
    }

    // per-tile row max (reduce across the 16 lanes holding one row)
    float mx[8];
#pragma unroll
    for (int r = 0; r < 8; ++r) {
      float m = l[r];
      m = fmaxf(m, __shfl_xor(m, 8, 16));
      m = fmaxf(m, __shfl_xor(m, 4, 16));
      m = fmaxf(m, __shfl_xor(m, 2, 16));
      m = fmaxf(m, __shfl_xor(m, 1, 16));
      mx[r] = m;
    }
    if (lcol == 0) {
#pragma unroll
      for (int r = 0; r < 8; ++r) wred[wave][r + half * 8] = mx[r];
    }
    __syncthreads();

    if (tid < 16) {
      float nm = rowmax[tid];
#pragma unroll
      for (int w = 0; w < 8; ++w) nm = fmaxf(nm, wred[w][tid]);
      const float sc = __expf(rowmax[tid] - nm);
      rowscale[tid] = sc;
      rowmax[tid]   = nm;
      rowsum[tid]  *= sc;
    }
    __syncthreads();

    // probabilities -> sP (bf16, A-frag friendly row-major), tile row sums
    float sm[8];
#pragma unroll
    for (int r = 0; r < 8; ++r) {
      const int M = r + half * 8;
      const float p = __expf(l[r] - rowmax[M]);
      sP[M * 128 + wave * 16 + lcol] = (bf16_t)p;
      sm[r] = p;
    }
#pragma unroll
    for (int r = 0; r < 8; ++r) {
      float s = sm[r];
      s += __shfl_xor(s, 8, 16);
      s += __shfl_xor(s, 4, 16);
      s += __shfl_xor(s, 2, 16);
      s += __shfl_xor(s, 1, 16);
      sm[r] = s;
    }
    if (lcol == 0) {
#pragma unroll
      for (int r = 0; r < 8; ++r) wred[wave][r + half * 8] = sm[r];
    }

    // rescale the running P@V accumulator by exp(old_max - new_max)
    {
      float sc[8];
#pragma unroll
      for (int r = 0; r < 8; ++r) sc[r] = rowscale[r + half * 8];
#pragma unroll
      for (int dt = 0; dt < 8; ++dt)
#pragma unroll
        for (int r = 0; r < 8; ++r) acc[dt][r] *= sc[r];
    }
    __syncthreads();

    if (tid < 16) {
      float s = rowsum[tid];
#pragma unroll
      for (int w = 0; w < 8; ++w) s += wred[w][tid];
      rowsum[tid] = s;
    }

    // ---- phase 2: acc += P @ V over this 128-wide j-block ------------------
    for (int kk = 0; kk < 4; ++kk) {
      ABFrag ap;
      const bf16_t* pp = &sP[lcol * 128 + kk * 32 + half * 8];
      ap.q[0] = *(const uint4*)(pp);
      ap.q[1] = *(const uint4*)(pp + 16);
#pragma unroll
      for (int dt = 0; dt < 8; ++dt) {
        const int d0 = wave * 128 + dt * 16;
        ABFrag bv;
        const bf16_t* vp =
            Vt + (size_t)(d0 + lcol) * NTOK + jb * 128 + kk * 32 + half * 8;
        bv.q[0] = *(const uint4*)(vp);
        bv.q[1] = *(const uint4*)(vp + 16);
        acc[dt] = wmma_bf16(ap.v, bv.v, acc[dt]);
      }
    }
    __syncthreads();
  }

  // ---- finalize: normalize + boundary-mask residual blend ------------------
  float inv[8], bm[8];
#pragma unroll
  for (int r = 0; r < 8; ++r) {
    const int M = r + half * 8;
    inv[r] = 1.0f / rowsum[M];
    bm[r]  = bmask[i0 + M];
  }
#pragma unroll
  for (int dt = 0; dt < 8; ++dt) {
    const int d = wave * 128 + dt * 16 + lcol;
#pragma unroll
    for (int r = 0; r < 8; ++r) {
      const int M = r + half * 8;
      const float xr = acc[dt][r] * inv[r];
      const float xv = x[(size_t)(i0 + M) * DDIM + d];
      out[(size_t)(i0 + M) * DDIM + d] = bm[r] * xr + (1.0f - bm[r]) * xv;
    }
  }
}

// ---------------------------------------------------------------------------
extern "C" void kernel_launch(void* const* d_in, const int* in_sizes, int n_in,
                              void* d_out, int out_size, void* d_ws, size_t ws_size,
                              hipStream_t stream) {
  const float* x    = (const float*)d_in[0];
  const float* am   = (const float*)d_in[1];
  const float* lm   = (const float*)d_in[2];
  const float* bmsk = (const float*)d_in[3];
  const float* Wq   = (const float*)d_in[4];
  const float* bq   = (const float*)d_in[5];
  const float* Wk   = (const float*)d_in[6];
  const float* bk   = (const float*)d_in[7];
  const float* Wv   = (const float*)d_in[8];
  const float* bv   = (const float*)d_in[9];
  const float* conn = (const float*)d_in[10];
  const float* bias = (const float*)d_in[11];
  float* out = (float*)d_out;

  // workspace layout (bf16 buffers, all resident in the 192MB L2)
  char* ws = (char*)d_ws;
  const size_t ND = (size_t)NTOK * DDIM * sizeof(bf16_t);  // 8 MB
  const size_t DD = (size_t)DDIM * DDIM * sizeof(bf16_t);  // 2 MB
  bf16_t* xb  = (bf16_t*)(ws);
  bf16_t* wqb = (bf16_t*)(ws + ND);
  bf16_t* wkb = (bf16_t*)(ws + ND + DD);
  bf16_t* wvb = (bf16_t*)(ws + ND + 2 * DD);
  bf16_t* cTb = (bf16_t*)(ws + ND + 3 * DD);
  bf16_t* Qb  = (bf16_t*)(ws + ND + 4 * DD);
  bf16_t* Kb  = (bf16_t*)(ws + 2 * ND + 4 * DD);
  bf16_t* Vt  = (bf16_t*)(ws + 3 * ND + 4 * DD);   // V transposed [D x N]
  bf16_t* Cb  = (bf16_t*)(ws + 4 * ND + 4 * DD);   // Q @ connection

  // 1) casts
  {
    int n4 = NTOK * DDIM / 4;
    cast_f32_bf16_v4<<<n4 / 256, 256, 0, stream>>>(x, xb, n4);
    int w4 = DDIM * DDIM / 4;
    cast_f32_bf16_v4<<<w4 / 256, 256, 0, stream>>>(Wq, wqb, w4);
    cast_f32_bf16_v4<<<w4 / 256, 256, 0, stream>>>(Wk, wkb, w4);
    cast_f32_bf16_v4<<<w4 / 256, 256, 0, stream>>>(Wv, wvb, w4);
    cast_transpose_bf16<<<(DDIM * DDIM) / 256, 256, 0, stream>>>(conn, cTb, DDIM, DDIM);
  }

  // 2) projections + 3) C = Q @ connection
  {
    dim3 grid(NTOK / 16, DDIM / 128);
    gemm16_bf16<<<grid, 256, 0, stream>>>(xb, wqb, bq, Qb, 0);
    gemm16_bf16<<<grid, 256, 0, stream>>>(xb, wkb, bk, Kb, 0);
    gemm16_bf16<<<grid, 256, 0, stream>>>(xb, wvb, bv, Vt, 1);
    gemm16_bf16<<<grid, 256, 0, stream>>>(Qb, cTb, nullptr, Cb, 0);
  }

  // 4) fused flash attention + straight-through mask + boundary blend
  flash_attn_st<<<NTOK / 16, 256, 0, stream>>>(Qb, Kb, Vt, Cb, am, lm, bmsk, x,
                                               bias, out);
}
